// GLformer_89343909691691
// MI455X (gfx1250) — matrix operand order repaired and checked by reading, not compiled, not run on hardware
//
#include <hip/hip_runtime.h>

#define BB 8
#define TT 2048
#define DD 512
#define HH 8
#define LL 512
#define DKK 64
#define KSZ 31

#define TM 128
#define TN 64
#define TK 32

typedef __attribute__((ext_vector_type(16))) __bf16 v16bf;
typedef __attribute__((ext_vector_type(8)))  __bf16 v8bf;
typedef __attribute__((ext_vector_type(8)))  float  v8f;

static __device__ __forceinline__ unsigned pk_bf16(float lo, float hi)
{
    __bf16 a = (__bf16)lo, b = (__bf16)hi;
    unsigned short ua = __builtin_bit_cast(unsigned short, a);
    unsigned short ub = __builtin_bit_cast(unsigned short, b);
    return ((unsigned)ub << 16) | (unsigned)ua;
}

// ---------------------------------------------------------------------------
// LDS-tiled, double-buffered strided/batched GEMM:
//   C = act(scale * (A x B) + bias [+ C_old])
// A element (z,m,k):  A + (z/zdiv)*az1 + (z%zdiv)*az2 + (m/mdiv)*am1 + (m%mdiv)*am2 + k
// B element (z,k,n):  B + (z/zdiv)*bz1 + (z%zdiv)*bz2 + k*bk + n*bn
// C element (z,m,n):  C + (z/zdiv)*cz1 + (z%zdiv)*cz2 + m*cm + n
// Requires M%16==0, N%64==0, K%32==0 (true at all call sites).
// Block = 256 threads (8 waves); block tile 128x64; wave tile 16x64 (4 WMMAs).
// ---------------------------------------------------------------------------
__global__ __launch_bounds__(256) void gemm_wmma_kernel(
    const float* __restrict__ A, const float* __restrict__ Bp,
    const float* __restrict__ bias, float* __restrict__ C,
    int M, int N, int K, int zdiv,
    long long az1, long long az2,
    int mdiv, long long am1, long long am2,
    long long bz1, long long bz2, long long bk, long long bn,
    long long cz1, long long cz2, long long cm,
    float scale, int beta, int act)
{
    __shared__ __attribute__((aligned(64))) __bf16 lA[2][TM * TK]; // [m][k] 2x8KB
    __shared__ __attribute__((aligned(64))) __bf16 lB[2][TN * TK]; // [n][k] 2x4KB

    const int tid  = threadIdx.x;
    const int lane = tid & 31;
    const int wave = tid >> 5;
    const int mBlock = blockIdx.y * TM;
    const int nBlock = blockIdx.x * TN;

    const int z = blockIdx.z;
    const long long zhi = z / zdiv, zlo = z % zdiv;
    const float* Az = A  + zhi * az1 + zlo * az2;
    const float* Bz = Bp + zhi * bz1 + zlo * bz2;
    float*       Cz = C  + zhi * cz1 + zlo * cz2;

    // --- cooperative global->LDS assignments (coalesced dword writes) ---
    // A: thread stages row mA = tid/2, k-pairs kpA0..kpA0+7 (128 rows x 16 pairs)
    const int mA   = tid >> 1;
    const int kpA0 = (tid & 1) * 8;
    const int mrA  = mBlock + mA;
    const bool aValid = mrA < M;
    const long long arowA = aValid
        ? (long long)(mrA / mdiv) * am1 + (long long)(mrA % mdiv) * am2 : 0;
    // B: thread stages column nB = tid/4, k-pairs kpB0..kpB0+3 (64 cols x 16 pairs)
    const int nB   = tid >> 2;
    const int kpB0 = (tid & 3) * 4;
    const long long bcol = (long long)(nBlock + nB) * bn;

    // --- wave compute setup (bf16 16x16x32 lanes, contiguous k = 16*(lane/16) + e) ---
    const int mW   = mBlock + wave * 16;
    const bool wValid = mW < M;
    const int fr   = lane & 15;              // fragment row (A) / col (B,C)
    const int akb  = (lane < 16) ? 0 : 16;   // A K base within 32
    const int bkb  = (lane < 16) ? 0 : 16;   // B K base within 32

    v8f acc[4];
#pragma unroll
    for (int j = 0; j < 4; ++j) acc[j] = v8f{0.f,0.f,0.f,0.f,0.f,0.f,0.f,0.f};

    // ---- staging lambda: pull one K-tile into LDS buffer `p` ----
    auto stage = [&](int kb, int p) {
        if (aValid) {
            const float* ap = Az + arowA + kb;
#pragma unroll
            for (int j = 0; j < 8; ++j) {
                const int kp = kpA0 + j;
                const float2 f = *(const float2*)(ap + kp * 2);
                *(unsigned*)&lA[p][mA * TK + kp * 2] = pk_bf16(f.x, f.y);
            }
            if (kb + TK < K) __builtin_prefetch(ap + TK, 0, 1);
        }
        if (bk == 1) {
            const float* bp = Bz + bcol + kb;
#pragma unroll
            for (int j = 0; j < 4; ++j) {
                const int kp = kpB0 + j;
                const float2 f = *(const float2*)(bp + kp * 2);
                *(unsigned*)&lB[p][nB * TK + kp * 2] = pk_bf16(f.x, f.y);
            }
            if (kb + TK < K) __builtin_prefetch(bp + TK, 0, 1);
        } else {
#pragma unroll
            for (int j = 0; j < 4; ++j) {
                const int kp = kpB0 + j;
                const long long k0 = kb + kp * 2;
                const float f0 = Bz[k0 * bk + bcol];
                const float f1 = Bz[(k0 + 1) * bk + bcol];
                *(unsigned*)&lB[p][nB * TK + kp * 2] = pk_bf16(f0, f1);
            }
            if (kb + TK < K)
                __builtin_prefetch(Bz + (long long)(kb + TK) * bk + bcol, 0, 1);
        }
    };

    int p = 0;
    stage(0, 0);
    for (int kb = 0; kb < K; kb += TK) {
        __syncthreads();                       // buffer p ready
        if (kb + TK < K) stage(kb + TK, p ^ 1); // overlap next stage with compute
        if (wValid) {
            // A fragment: two contiguous 16B runs per lane
            const __bf16* ar = &lA[p][(wave * 16 + fr) * TK];
            const v8bf alo = *(const v8bf*)(ar + akb);
            const v8bf ahi = *(const v8bf*)(ar + akb + 8);
            v16bf a;
#pragma unroll
            for (int e = 0; e < 8; ++e) { a[e] = alo[e]; a[8 + e] = ahi[e]; }
            // preload all four B fragments (one contiguous 32B run each),
            // then issue the four WMMAs back-to-back
            v16bf bf[4];
#pragma unroll
            for (int j = 0; j < 4; ++j)
                bf[j] = *(const v16bf*)&lB[p][(j * 16 + fr) * TK + bkb];
#pragma unroll
            for (int j = 0; j < 4; ++j)
                acc[j] = __builtin_amdgcn_wmma_f32_16x16x32_bf16(
                             false, a, false, bf[j], (short)0, acc[j], false, false);
        }
        p ^= 1;
    }

    if (!wValid) return;
#pragma unroll
    for (int j = 0; j < 4; ++j) {
        const int nc = nBlock + j * 16 + fr;
#pragma unroll
        for (int r = 0; r < 8; ++r) {
            const int mm = mW + r + ((lane < 16) ? 0 : 8);
            float v = acc[j][r] * scale;
            if (bias != nullptr) v += bias[nc];
            const long long co = (long long)mm * cm + nc;
            if (beta) v += Cz[co];
            if (act == 1) v = 1.0f / (1.0f + __expf(-v));
            Cz[co] = v;
        }
    }
}

// ---------------------------------------------------------------------------
// LayerNorm over last dim (one block per row)
// ---------------------------------------------------------------------------
__global__ __launch_bounds__(256) void layernorm_kernel(
    const float* __restrict__ x, const float* __restrict__ g,
    const float* __restrict__ b, float* __restrict__ y, int Dn)
{
    __shared__ float red[256];
    const long long row = blockIdx.x;
    const float* xr = x + row * Dn;
    float s = 0.f;
    for (int i = threadIdx.x; i < Dn; i += 256) s += xr[i];
    red[threadIdx.x] = s; __syncthreads();
    for (int st = 128; st > 0; st >>= 1) {
        if (threadIdx.x < st) red[threadIdx.x] += red[threadIdx.x + st];
        __syncthreads();
    }
    const float mean = red[0] / Dn;
    __syncthreads();
    float v = 0.f;
    for (int i = threadIdx.x; i < Dn; i += 256) {
        const float d = xr[i] - mean; v += d * d;
    }
    red[threadIdx.x] = v; __syncthreads();
    for (int st = 128; st > 0; st >>= 1) {
        if (threadIdx.x < st) red[threadIdx.x] += red[threadIdx.x + st];
        __syncthreads();
    }
    const float rstd = rsqrtf(red[0] / Dn + 1e-5f);
    float* yr = y + row * Dn;
    for (int i = threadIdx.x; i < Dn; i += 256)
        yr[i] = (xr[i] - mean) * rstd * g[i] + b[i];
}

// ---------------------------------------------------------------------------
// Row softmax (in place), one block per row
// ---------------------------------------------------------------------------
__global__ __launch_bounds__(256) void softmax_kernel(float* __restrict__ s, int W)
{
    __shared__ float red[256];
    float* r = s + (long long)blockIdx.x * W;
    float mx = -3.4e38f;
    for (int i = threadIdx.x; i < W; i += 256) mx = fmaxf(mx, r[i]);
    red[threadIdx.x] = mx; __syncthreads();
    for (int st = 128; st > 0; st >>= 1) {
        if (threadIdx.x < st) red[threadIdx.x] = fmaxf(red[threadIdx.x], red[threadIdx.x + st]);
        __syncthreads();
    }
    mx = red[0];
    __syncthreads();
    float sum = 0.f;
    for (int i = threadIdx.x; i < W; i += 256) {
        const float e = __expf(r[i] - mx); r[i] = e; sum += e;
    }
    red[threadIdx.x] = sum; __syncthreads();
    for (int st = 128; st > 0; st >>= 1) {
        if (threadIdx.x < st) red[threadIdx.x] += red[threadIdx.x + st];
        __syncthreads();
    }
    const float inv = 1.0f / red[0];
    for (int i = threadIdx.x; i < W; i += 256) r[i] *= inv;
}

// Average-pool by 4 along time: out[r,d] = mean(x[4r..4r+3, d])
__global__ void pool4_kernel(const float* __restrict__ x, float* __restrict__ y,
                             long long n, int Dn)
{
    const long long i = (long long)blockIdx.x * blockDim.x + threadIdx.x;
    if (i >= n) return;
    const long long r = i / Dn;
    const int d = (int)(i % Dn);
    const float* p = x + (r * 4) * Dn + d;
    y[i] = 0.25f * (p[0] + p[Dn] + p[2LL * Dn] + p[3LL * Dn]);
}

// GLU: y[r,c] = h[r,c] * sigmoid(h[r,c+half])
__global__ void glu_kernel(const float* __restrict__ h, float* __restrict__ y,
                           long long rows, int half)
{
    const long long i = (long long)blockIdx.x * blockDim.x + threadIdx.x;
    if (i >= rows * half) return;
    const long long r = i / half;
    const int c = (int)(i % half);
    const float a = h[r * (2LL * half) + c];
    const float g = h[r * (2LL * half) + half + c];
    y[i] = a / (1.0f + __expf(-g));
}

// out = xin + gate * up4(ysmall)   (gate already sigmoided)
__global__ void gated_up_res_kernel(const float* __restrict__ xin,
                                    const float* __restrict__ gate,
                                    const float* __restrict__ ysmall,
                                    float* __restrict__ out,
                                    long long n, int Dn, int Tn)
{
    const long long i = (long long)blockIdx.x * blockDim.x + threadIdx.x;
    if (i >= n) return;
    const int d = (int)(i % Dn);
    const long long bt = i / Dn;
    const long long t = bt % Tn;
    const long long b = bt / Tn;
    const long long si = (b * (Tn / 4) + (t >> 2)) * Dn + d;
    out[i] = xin[i] + gate[i] * ysmall[si];
}

__global__ void add_kernel(const float* __restrict__ a, const float* __restrict__ b,
                           float* __restrict__ o, long long n)
{
    const long long i = (long long)blockIdx.x * blockDim.x + threadIdx.x;
    if (i < n) o[i] = a[i] + b[i];
}

// Depthwise conv along t, kernel 31, SAME padding
__global__ void dwconv_kernel(const float* __restrict__ y, const float* __restrict__ w,
                              const float* __restrict__ bias, float* __restrict__ o,
                              int Bn, int Tn, int Dn)
{
    const long long i = (long long)blockIdx.x * blockDim.x + threadIdx.x;
    const long long total = (long long)Bn * Tn * Dn;
    if (i >= total) return;
    const int d = (int)(i % Dn);
    const long long bt = i / Dn;
    const int t = (int)(bt % Tn);
    const long long b = bt / Tn;
    float acc = bias[d];
    const float* wr = w + (long long)d * KSZ;
#pragma unroll
    for (int k = 0; k < KSZ; ++k) {
        const int tt = t + k - (KSZ / 2);
        if (tt >= 0 && tt < Tn) acc += y[(b * Tn + tt) * Dn + d] * wr[k];
    }
    o[i] = acc;
}

// ---------------------------------------------------------------------------
// Host-side helpers
// ---------------------------------------------------------------------------
static inline void gemm(hipStream_t st, const float* A, const float* B,
                        const float* bias, float* C,
                        int M, int N, int K, int batch, int zdiv,
                        long long az1, long long az2,
                        int mdiv, long long am1, long long am2,
                        long long bz1, long long bz2, long long bk, long long bn,
                        long long cz1, long long cz2, long long cm,
                        float scale, int beta, int act)
{
    dim3 grid(N / TN, (M + TM - 1) / TM, batch);
    gemm_wmma_kernel<<<grid, 256, 0, st>>>(A, B, bias, C, M, N, K, zdiv,
                                           az1, az2, mdiv, am1, am2,
                                           bz1, bz2, bk, bn, cz1, cz2, cm,
                                           scale, beta, act);
}

// Plain row-major: C[M,N] = act(A[M,K] @ B[K,N] + bias)
static inline void gemm_rm(hipStream_t st, const float* A, const float* B,
                           const float* bias, float* C, int M, int N, int K, int act)
{
    gemm(st, A, B, bias, C, M, N, K, /*batch*/1, /*zdiv*/1,
         0, 0, /*mdiv*/1, (long long)K, 0,
         0, 0, (long long)N, 1,
         0, 0, (long long)N,
         1.0f, 0, act);
}

static inline long long ceildiv(long long a, long long b) { return (a + b - 1) / b; }

extern "C" void kernel_launch(void* const* d_in, const int* in_sizes, int n_in,
                              void* d_out, int out_size, void* d_ws, size_t ws_size,
                              hipStream_t stream)
{
    (void)in_sizes; (void)n_in; (void)out_size; (void)ws_size;
    const float* x      = (const float*)d_in[0];
    const float* pos_k  = (const float*)d_in[1];
    const float* mha_g  = (const float*)d_in[2];
    const float* mha_b  = (const float*)d_in[3];
    const float* wq     = (const float*)d_in[4];
    const float* bq     = (const float*)d_in[5];
    const float* wk     = (const float*)d_in[6];
    const float* bk     = (const float*)d_in[7];
    const float* wv     = (const float*)d_in[8];
    const float* bv     = (const float*)d_in[9];
    const float* wo     = (const float*)d_in[10];
    const float* bo     = (const float*)d_in[11];
    const float* ega_wg = (const float*)d_in[12];
    const float* ega_bg = (const float*)d_in[13];
    const float* g1_g   = (const float*)d_in[14];
    const float* g1_b   = (const float*)d_in[15];
    const float* g1_w1  = (const float*)d_in[16];
    const float* g1_b1  = (const float*)d_in[17];
    const float* g1_w2  = (const float*)d_in[18];
    const float* g1_b2  = (const float*)d_in[19];
    const float* g1_wg  = (const float*)d_in[20];
    const float* g1_bg  = (const float*)d_in[21];
    const float* cla_g  = (const float*)d_in[22];
    const float* cla_b  = (const float*)d_in[23];
    const float* cla_w1 = (const float*)d_in[24];
    const float* cla_b1 = (const float*)d_in[25];
    const float* cla_dw = (const float*)d_in[26];
    const float* cla_db = (const float*)d_in[27];
    const float* cla_w2 = (const float*)d_in[28];
    const float* cla_b2 = (const float*)d_in[29];
    const float* g2_g   = (const float*)d_in[30];
    const float* g2_b   = (const float*)d_in[31];
    const float* g2_w1  = (const float*)d_in[32];
    const float* g2_b1  = (const float*)d_in[33];
    const float* g2_w2  = (const float*)d_in[34];
    const float* g2_b2  = (const float*)d_in[35];
    const float* g2_wg  = (const float*)d_in[36];
    const float* g2_bg  = (const float*)d_in[37];

    float* ws = (float*)d_ws;
    const long long SZ_FULL  = (long long)BB * TT * DD;          // 8,388,608
    const long long SZ_SMALL = (long long)BB * (TT / 4) * DD;    // 2,097,152
    const long long SZ_BIG   = (long long)BB * HH * LL * LL;     // 16,777,216

    long long off = 0;
    float* X  = ws + off; off += SZ_FULL;
    float* SA = ws + off; off += SZ_SMALL;
    float* SB = ws + off; off += SZ_SMALL;
    float* Q  = ws + off; off += SZ_SMALL;
    float* Kt = ws + off; off += SZ_SMALL;
    float* V  = ws + off; off += SZ_SMALL;
    float* O  = ws + off; off += SZ_SMALL;
    float* S  = ws + off; off += SZ_BIG;    // scores; reused as GCFN/CLA hidden
    float* H1 = S;
    float* XG = ws + off; off += SZ_FULL;
    float* Y1 = ws + off; off += SZ_FULL;
    float* Y2 = ws + off; off += SZ_FULL;

    const int RT = BB * TT;        // 16384 full rows
    const int R4 = BB * (TT / 4);  // 4096 pooled rows
    hipStream_t st = stream;

    // ================= EGA =================
    pool4_kernel<<<(unsigned)ceildiv(SZ_SMALL, 256), 256, 0, st>>>(x, SA, SZ_SMALL, DD);
    layernorm_kernel<<<R4, 256, 0, st>>>(SA, mha_g, mha_b, SB, DD);
    gemm_rm(st, SB, wq, bq, Q,  R4, DD, DD, 0);
    gemm_rm(st, SB, wk, bk, Kt, R4, DD, DD, 0);
    gemm_rm(st, SB, wv, bv, V,  R4, DD, DD, 0);

    // S[b,h,i,j]  = 0.125 * sum_d q[b,i,h,d] * pos_k[i,j,d]   (batch over i)
    gemm(st, Q, pos_k, nullptr, S, /*M*/BB * HH, /*N*/LL, /*K*/DKK, /*batch*/LL, /*zdiv*/1,
         /*az1*/512, /*az2*/0,
         /*mdiv*/HH, /*am1*/(long long)LL * 512, /*am2*/64,
         /*bz1*/(long long)LL * DKK, /*bz2*/0, /*bk*/1, /*bn*/DKK,
         /*cz1*/LL, /*cz2*/0, /*cm*/(long long)LL * LL,
         0.125f, 0, 0);

    // S[b,h,i,j] += 0.125 * sum_d q[b,i,h,d] * k[b,j,h,d]     (batch over b*h)
    gemm(st, Q, Kt, nullptr, S, /*M*/LL, /*N*/LL, /*K*/DKK, /*batch*/BB * HH, /*zdiv*/HH,
         (long long)LL * 512, 64,
         1, 512, 0,
         (long long)LL * 512, 64, 1, 512,
         (long long)HH * LL * LL, (long long)LL * LL, LL,
         0.125f, 1, 0);

    softmax_kernel<<<BB * HH * LL, 256, 0, st>>>(S, LL);

    // O[b,i,h,d] = sum_j attn[b,h,i,j] * v[b,j,h,d]
    gemm(st, S, V, nullptr, O, /*M*/LL, /*N*/DKK, /*K*/LL, /*batch*/BB * HH, /*zdiv*/HH,
         (long long)HH * LL * LL, (long long)LL * LL,
         1, LL, 0,
         (long long)LL * 512, 64, 512, 1,
         (long long)LL * 512, 64, 512,
         1.0f, 0, 0);

    gemm_rm(st, O, wo, bo, SB, R4, DD, DD, 0);                  // O2 -> SB
    gemm_rm(st, x, ega_wg, ega_bg, XG, RT, DD, DD, 1);          // sigmoid gate
    gated_up_res_kernel<<<(unsigned)ceildiv(SZ_FULL, 256), 256, 0, st>>>(
        x, XG, SB, X, SZ_FULL, DD, TT);

    // ================= GCFN #1 (global) =================
    pool4_kernel<<<(unsigned)ceildiv(SZ_SMALL, 256), 256, 0, st>>>(X, SA, SZ_SMALL, DD);
    layernorm_kernel<<<R4, 256, 0, st>>>(SA, g1_g, g1_b, SB, DD);
    gemm_rm(st, SB, g1_w1, g1_b1, H1, R4, 6 * DD, DD, 0);
    glu_kernel<<<(unsigned)ceildiv((long long)R4 * 3 * DD, 256), 256, 0, st>>>(
        H1, Y1, R4, 3 * DD);
    gemm_rm(st, Y1, g1_w2, g1_b2, Y2, R4, DD, 3 * DD, 0);
    gemm_rm(st, X, g1_wg, g1_bg, XG, RT, DD, DD, 1);
    gated_up_res_kernel<<<(unsigned)ceildiv(SZ_FULL, 256), 256, 0, st>>>(
        X, XG, Y2, X, SZ_FULL, DD, TT);

    // ================= CLA =================
    layernorm_kernel<<<RT, 256, 0, st>>>(X, cla_g, cla_b, XG, DD);   // LN out -> XG
    gemm_rm(st, XG, cla_w1, cla_b1, H1, RT, 2 * DD, DD, 0);
    glu_kernel<<<(unsigned)ceildiv((long long)RT * DD, 256), 256, 0, st>>>(
        H1, Y1, RT, DD);
    dwconv_kernel<<<(unsigned)ceildiv(SZ_FULL, 256), 256, 0, st>>>(
        Y1, cla_dw, cla_db, Y2, BB, TT, DD);
    gemm_rm(st, Y2, cla_w2, cla_b2, Y1, RT, DD, DD, 0);
    add_kernel<<<(unsigned)ceildiv(SZ_FULL, 256), 256, 0, st>>>(X, Y1, X, SZ_FULL);

    // ================= GCFN #2 (local) =================
    pool4_kernel<<<(unsigned)ceildiv(SZ_SMALL, 256), 256, 0, st>>>(X, SA, SZ_SMALL, DD);
    layernorm_kernel<<<R4, 256, 0, st>>>(SA, g2_g, g2_b, SB, DD);
    gemm_rm(st, SB, g2_w1, g2_b1, H1, R4, 6 * DD, DD, 0);
    glu_kernel<<<(unsigned)ceildiv((long long)R4 * 3 * DD, 256), 256, 0, st>>>(
        H1, Y1, R4, 3 * DD);
    gemm_rm(st, Y1, g2_w2, g2_b2, Y2, R4, DD, 3 * DD, 0);
    gemm_rm(st, X, g2_wg, g2_bg, XG, RT, DD, DD, 1);
    gated_up_res_kernel<<<(unsigned)ceildiv(SZ_FULL, 256), 256, 0, st>>>(
        X, XG, Y2, (float*)d_out, SZ_FULL, DD, TT);
}
